// ScannedRNN_21672404976167
// MI455X (gfx1250) — matrix operand order repaired and checked by reading, not compiled, not run on hardware
//
#include <hip/hip_runtime.h>
#include <hip/hip_bf16.h>

#define T_STEPS 512
#define BATCH   128
#define DDIM    512
#define HDIM    512
#define G4H     2048
#define LDS_PAD 16
#define XROW    (DDIM + LDS_PAD)   // 528 bf16 = 1056 B row stride (16B aligned)

typedef __bf16 bf16;
typedef __attribute__((ext_vector_type(16))) __bf16 v16bf;
typedef __attribute__((ext_vector_type(8)))  __bf16 bf16x8;
typedef __attribute__((ext_vector_type(8)))  float  v8f;
typedef __attribute__((ext_vector_type(4)))  float  f32x4;

// ---------------------------------------------------------------------------
// Prep 1: swizzle fp32 weight [512 x 2048] (row-major) into bf16 WMMA
// B-fragment layout.  Tile = 32(K) x 16(N).  tile = kt*128 + ntile, within
// tile lane*16 + j contiguous bf16:
//   lane <  16 : N = lane,      K = j        (j = 0..15)
//   lane >= 16 : N = lane - 16, K = 16 + j
// ---------------------------------------------------------------------------
__global__ void swizzle_weights(const float* __restrict__ W, bf16* __restrict__ out) {
  int e = blockIdx.x * blockDim.x + threadIdx.x;   // 0 .. 16*128*512-1
  int kt    = e >> 16;
  int rem   = e & 65535;
  int ntile = rem >> 9;
  int q     = rem & 511;
  int lane  = q >> 4;
  int j     = q & 15;
  int nn = lane & 15;
  int kk = (lane < 16) ? j : (16 + j);
  out[e] = (bf16)W[(kt * 32 + kk) * G4H + ntile * 16 + nn];
}

// Prep 2: ins fp32 -> bf16 (same [T,B,D] layout) so the scan kernel can DMA
// rows straight into LDS with async-to-LDS copies (no per-step cvt on the
// serial path).
__global__ void convert_ins(const float* __restrict__ in, bf16* __restrict__ out) {
  size_t i = ((size_t)blockIdx.x * blockDim.x + threadIdx.x) * 4;
  f32x4 v = *(const f32x4*)(in + i);
#pragma unroll
  for (int k = 0; k < 4; ++k) out[i + k] = (bf16)v[k];
}

__device__ __forceinline__ float sigmoid_fast(float x) {
  return 1.0f / (1.0f + __expf(-x));
}
__device__ __forceinline__ float tanh_fast(float x) {
  return 2.0f / (1.0f + __expf(-2.0f * x)) - 1.0f;
}

// A-fragment (16x32 bf16) from an LDS row:
//   lane < 16 : M=lane,    K = {0..15}   -> half8 = 0
//   lane >=16 : M=lane-16, K = {16..31}  -> half8 = 8
__device__ __forceinline__ v16bf load_a(const bf16* rowp, int kbase, int half8) {
  bf16x8 lo = *(const bf16x8*)(rowp + kbase + 2 * half8);
  bf16x8 hi = *(const bf16x8*)(rowp + kbase + 2 * half8 + 8);
  v16bf r;
#pragma unroll
  for (int i = 0; i < 8; ++i) { r[i] = lo[i]; r[i + 8] = hi[i]; }
  return r;
}

// B-fragment: pre-swizzled, 32 contiguous bytes per lane.
__device__ __forceinline__ v16bf load_b(const bf16* __restrict__ wt, int kt, int ntile, int lane) {
  const bf16* p = wt + (((kt << 7) + ntile) << 9) + (lane << 4);
  bf16x8 lo = *(const bf16x8*)(p);
  bf16x8 hi = *(const bf16x8*)(p + 8);
  v16bf r;
#pragma unroll
  for (int i = 0; i < 8; ++i) { r[i] = lo[i]; r[i + 8] = hi[i]; }
  return r;
}

// ---------------------------------------------------------------------------
// Persistent scan kernel. 1 block = 16 batch rows for all T steps.
// 512 threads = 16 waves; wave w owns hidden columns [32w, 32w+32) of every
// gate (ntile = g*32 + 2w + tt) so the cell update is entirely in-wave.
// x_{t+1} is DMA'd into a double-buffered LDS tile with async-to-LDS copies
// issued a full step ahead (ASYNCcnt), so staging never blocks the scan.
// ---------------------------------------------------------------------------
__global__ void __launch_bounds__(512, 1)
lstm_scan(const bf16* __restrict__ insb,
          const unsigned char* __restrict__ resets,
          const float* __restrict__ c0,
          const float* __restrict__ h0,
          const bf16* __restrict__ WiT,
          const bf16* __restrict__ WhT,
          const float* __restrict__ bias,
          float* __restrict__ out) {
  const int b0    = blockIdx.x * 16;
  const int tid   = threadIdx.x;
  const int wave  = tid >> 5;
  const int lane  = tid & 31;
  const int nn    = lane & 15;
  const int half8 = (lane < 16) ? 0 : 8;  // A-frag K offset
  const int rowoff= (lane < 16) ? 0 : 8;  // C/D row offset (M = r + rowoff)

  __shared__ bf16 x_tiles[2][16][XROW];
  __shared__ bf16 h_tile[16][XROW];
  __shared__ unsigned char s_reset[16];

  // staging mapping (h0 + reset-zeroing): thread owns 16 cols of one row
  const int srow = tid >> 5;
  const int scol = (tid & 31) << 4;

  // async DMA of one x row (1 KB) per wave: 2 x B128 (16 B/lane, 512 B/op).
  // ISA applies INST_OFFSET to both LDS and global addresses.
  auto issue_async_x = [&](int tstep, int buf) {
    const bf16* src = insb + ((size_t)tstep * BATCH + b0 + wave) * DDIM;
    unsigned l0 = (unsigned)(size_t)(void*)&x_tiles[buf][wave][0] + (unsigned)(lane << 4);
    unsigned long long g0 =
        (unsigned long long)(size_t)src + (unsigned long long)(lane << 4);
    asm volatile("global_load_async_to_lds_b128 %0, %1, off"
                 :: "v"(l0), "v"(g0) : "memory");
    asm volatile("global_load_async_to_lds_b128 %0, %1, off offset:512"
                 :: "v"(l0), "v"(g0) : "memory");
  };

  // stage h0 -> LDS (bf16)
#pragma unroll
  for (int i = 0; i < 16; i += 4) {
    f32x4 v = *(const f32x4*)(h0 + (size_t)(b0 + srow) * HDIM + scol + i);
#pragma unroll
    for (int k = 0; k < 4; ++k) h_tile[srow][scol + i + k] = (bf16)v[k];
  }

  // persistent cell state + bias per accumulator
  v8f c_reg[2], h_out[2];
  float bb[8];
#pragma unroll
  for (int tt = 0; tt < 2; ++tt) {
    int col = (wave << 5) + (tt << 4) + nn;
#pragma unroll
    for (int r = 0; r < 8; ++r)
      c_reg[tt][r] = c0[(size_t)(b0 + r + rowoff) * HDIM + col];
  }
#pragma unroll
  for (int g = 0; g < 4; ++g)
#pragma unroll
    for (int tt = 0; tt < 2; ++tt)
      bb[g * 2 + tt] = bias[g * 512 + ((wave * 2 + tt) << 4) + nn];

  // prime the pipeline: DMA x_0 into buffer 0
  issue_async_x(0, 0);

  float* ys_base = out + 2 * (size_t)BATCH * HDIM;

  for (int t = 0; t < T_STEPS; ++t) {
    const int p = t & 1;
    asm volatile("s_wait_asynccnt 0" ::: "memory");  // own DMA into buf p done
    __syncthreads();                 // all waves' DMA done; prev h writes done

    if (t + 1 < T_STEPS) issue_async_x(t + 1, p ^ 1);  // hide behind full step
    if (tid < 16) s_reset[tid] = resets[t * BATCH + b0 + tid];
    __syncthreads();                 // s_reset visible

    // reset masking: h rows in LDS (consumed only after barrier C), c in regs
    if (s_reset[srow]) {
#pragma unroll
      for (int i = 0; i < 16; ++i) h_tile[srow][scol + i] = (bf16)0.0f;
    }
#pragma unroll
    for (int tt = 0; tt < 2; ++tt)
#pragma unroll
      for (int r = 0; r < 8; ++r)
        if (s_reset[r + rowoff]) c_reg[tt][r] = 0.0f;

    // z = x@Wi + h@Wh + b   (bf16 WMMA, f32 accumulate)
    v8f acc[8];
#pragma unroll
    for (int a = 0; a < 8; ++a)
#pragma unroll
      for (int r = 0; r < 8; ++r) acc[a][r] = bb[a];

    // x-GEMM first: independent of h masking, hides the mask + barrier
#pragma unroll 2
    for (int kt = 0; kt < 16; ++kt) {
      v16bf af = load_a(&x_tiles[p][lane & 15][0], kt << 5, half8);
#pragma unroll
      for (int g = 0; g < 4; ++g)
#pragma unroll
        for (int tt = 0; tt < 2; ++tt) {
          v16bf bw = load_b(WiT, kt, g * 32 + wave * 2 + tt, lane);
          acc[g * 2 + tt] = __builtin_amdgcn_wmma_f32_16x16x32_bf16(
              false, af, false, bw, (short)0, acc[g * 2 + tt], false, false);
        }
    }
    __syncthreads();                 // masked h visible

#pragma unroll 2
    for (int kt = 0; kt < 16; ++kt) {
      v16bf af = load_a(&h_tile[lane & 15][0], kt << 5, half8);
#pragma unroll
      for (int g = 0; g < 4; ++g)
#pragma unroll
        for (int tt = 0; tt < 2; ++tt) {
          v16bf bw = load_b(WhT, kt, g * 32 + wave * 2 + tt, lane);
          acc[g * 2 + tt] = __builtin_amdgcn_wmma_f32_16x16x32_bf16(
              false, af, false, bw, (short)0, acc[g * 2 + tt], false, false);
        }
    }
    __syncthreads();                 // everyone done reading h_tile

    // gates + cell update; write new h to LDS (bf16) and ys (fp32)
    float* ys = ys_base + (size_t)t * BATCH * HDIM;
#pragma unroll
    for (int tt = 0; tt < 2; ++tt) {
      int col = (wave << 5) + (tt << 4) + nn;
#pragma unroll
      for (int r = 0; r < 8; ++r) {
        float iv = sigmoid_fast(acc[0 + tt][r]);
        float fv = sigmoid_fast(acc[2 + tt][r]);
        float gv = tanh_fast(acc[4 + tt][r]);
        float ov = sigmoid_fast(acc[6 + tt][r]);
        float nc = fv * c_reg[tt][r] + iv * gv;
        float nh = ov * tanh_fast(nc);
        c_reg[tt][r] = nc;
        h_out[tt][r] = nh;
        h_tile[r + rowoff][col] = (bf16)nh;
        ys[(size_t)(b0 + r + rowoff) * HDIM + col] = nh;
      }
    }
  }

  // final cT, hT
#pragma unroll
  for (int tt = 0; tt < 2; ++tt) {
    int col = (wave << 5) + (tt << 4) + nn;
#pragma unroll
    for (int r = 0; r < 8; ++r) {
      out[(size_t)(b0 + r + rowoff) * HDIM + col] = c_reg[tt][r];
      out[(size_t)BATCH * HDIM + (size_t)(b0 + r + rowoff) * HDIM + col] = h_out[tt][r];
    }
  }
}

extern "C" void kernel_launch(void* const* d_in, const int* in_sizes, int n_in,
                              void* d_out, int out_size, void* d_ws, size_t ws_size,
                              hipStream_t stream) {
  const float*         ins    = (const float*)d_in[0];
  const unsigned char* resets = (const unsigned char*)d_in[1];  // jax bool = 1 byte
  const float*         c0     = (const float*)d_in[2];
  const float*         h0     = (const float*)d_in[3];
  const float*         Wi     = (const float*)d_in[4];
  const float*         Wh     = (const float*)d_in[5];
  const float*         b      = (const float*)d_in[6];
  float*               out    = (float*)d_out;

  bf16* insB = (bf16*)d_ws;                                   // 64 MB
  bf16* WiT  = insB + (size_t)T_STEPS * BATCH * DDIM;         // 2 MB
  bf16* WhT  = WiT + (size_t)16 * 128 * 512;                  // 2 MB

  // one-time preps (fully parallel, off the serial path)
  convert_ins<<<32768, 256, 0, stream>>>(ins, insB);          // 33.5M elems / 4 per thread
  swizzle_weights<<<4096, 256, 0, stream>>>(Wi, WiT);
  swizzle_weights<<<4096, 256, 0, stream>>>(Wh, WhT);

  // persistent scan: 8 blocks (16 batch rows each) x 512 threads (16 waves)
  lstm_scan<<<BATCH / 16, 512, 0, stream>>>(insB, resets, c0, h0, WiT, WhT, b, out);
}
